// MultiHeadedAttention_31945966747768
// MI455X (gfx1250) — matrix operand order repaired and checked by reading, not compiled, not run on hardware
//
#include <hip/hip_runtime.h>
#include <hip/hip_bf16.h>

// MAT-style multi-headed attention with edge features, fused for MI455X (gfx1250).
// eK/eV ([B,N,N,H,DK] = 268MB each) are never materialized: only hiddenK/hiddenV
// ([B,N,N,64] bf16 = 8MB, L2-resident) are kept, with eK_W2/eV_W2 folded into small
// per-head 64x64 GEMMs. Row-constant score terms (k.rel_u, eK_b2 terms) cancel in
// softmax and are dropped. All matmuls: v_wmma_f32_16x16x32_bf16, fp32 accumulate.
// All fp32 operands are converted to bf16 exactly once in a streaming prep kernel,
// in WMMA-tile-friendly layouts, so every hot GEMM loop is 128-bit loads + wmma.
//
// Weight tile layout ("wtile"): for K-block kb (32 rows) and column n, the 16 dwords
//   Wt[(kb*1024 + n)*16 + hi*8 + p]  hold bf16 pairs (W[k,n],W[k+1,n]), k = kb*32+hi*16+2p.
// Lane half h's WMMA B-operand is dwords 4h+0..3 and 4h+8..11 of these 16 (2x b128 loads).

typedef __bf16 bf16;
typedef __attribute__((ext_vector_type(16))) __bf16 v16bf;
typedef __attribute__((ext_vector_type(8)))  float  v8f;
typedef __attribute__((ext_vector_type(8)))  unsigned v8u;

#define DEV __device__ __forceinline__

DEV unsigned short f2bf_bits(float f) {
  unsigned u = __builtin_bit_cast(unsigned, f);
  u += 0x7FFFu + ((u >> 16) & 1u);              // round-to-nearest-even
  return (unsigned short)(u >> 16);
}
// pack two floats into one dword of bf16: lo -> [15:0], hi -> [31:16]
DEV unsigned pack2(float lo, float hi) {
  unsigned ul = __builtin_bit_cast(unsigned, lo);
  unsigned uh = __builtin_bit_cast(unsigned, hi);
  ul += 0x7FFFu + ((ul >> 16) & 1u);
  uh += 0x7FFFu + ((uh >> 16) & 1u);
  return (uh & 0xFFFF0000u) | (ul >> 16);
}
// (bf16,bf16)+(bf16,bf16) elementwise in fp32, repacked
DEV unsigned addpack_bf(unsigned x, unsigned y) {
  float lx = __builtin_bit_cast(float, x << 16);
  float hx = __builtin_bit_cast(float, x & 0xFFFF0000u);
  float ly = __builtin_bit_cast(float, y << 16);
  float hy = __builtin_bit_cast(float, y & 0xFFFF0000u);
  return pack2(lx + ly, hx + hy);
}

DEV v8f wmma_bf16(v16bf a, v16bf b, v8f c) {
  return __builtin_amdgcn_wmma_f32_16x16x32_bf16(false, a, false, b, (short)0, c, false, false);
}

// WMMA 16x16x32 bf16 operand maps (CDNA5 ISA 7.12.2, wave32):
//  A (16x32): lane M=lane&15; element e -> K = 8*(lane>>4) + e + (e<8?0:8)
//    => dwords: u[0..3] at K-dword base+0..3, u[4..7] at base+8..11
//  B (32x16): lane N=lane&15; element e -> K = 8*(lane>>4) + e + (e<8?0:8), same dword stride as A
//  C/D (16x16 f32): vgpr v -> M = v + 8*(lane>>4), N = lane&15
DEV v16bf load16_contig(const unsigned* p) {
  v8u u;
#pragma unroll
  for (int i = 0; i < 8; ++i) u[i] = p[i];
  return __builtin_bit_cast(v16bf, u);
}
DEV v16bf load16_astride(const unsigned* p) {
  v8u u;
#pragma unroll
  for (int i = 0; i < 4; ++i) { u[i] = p[i]; u[i + 4] = p[i + 8]; }
  return __builtin_bit_cast(v16bf, u);
}

#define LANE_SETUP() \
  const int lane = threadIdx.x & 31; const int hi = lane >> 4; \
  const int am = lane & 15; const int akb = hi * 8;            \
  const int bn = am;        const int bkb = hi * 16;           \
  (void)am; (void)akb; (void)bn; (void)bkb;
#define WID() (blockIdx.x * (blockDim.x >> 5) + (threadIdx.x >> 5))

// ---------------------------------------------------------------------------
// 0) One-pass bf16 conversion of all fp32 operands into WMMA-friendly layouts.
//    - QKVO weights (1024x1024) & W2 (64x1024): wtile layout (see header)
//    - activations q/k/v (512x1024): row-major bf16 (pairs contiguous in k)
//    - edges [B,E,N,N] -> transposed [b,i,j,e] bf16 (pairs contiguous in e)
//    - W1 (16x64): per-column 8 contiguous dwords W1t[c*8+p] = (W1[2p,c],W1[2p+1,c])
__global__ void k_prep(const float* __restrict__ Wq, const float* __restrict__ Wk,
                       const float* __restrict__ Wv, const float* __restrict__ Wo,
                       const float* __restrict__ W2K, const float* __restrict__ W2V,
                       const float* __restrict__ Xq, const float* __restrict__ Xk,
                       const float* __restrict__ Xv, const float* __restrict__ edges,
                       const float* __restrict__ W1K, const float* __restrict__ W1V,
                       unsigned* __restrict__ Wqp, unsigned* __restrict__ Wkp,
                       unsigned* __restrict__ Wvp, unsigned* __restrict__ Wop,
                       unsigned* __restrict__ W2Kp, unsigned* __restrict__ W2Vp,
                       unsigned* __restrict__ Xqd, unsigned* __restrict__ Xkd,
                       unsigned* __restrict__ Xvd, unsigned* __restrict__ Ed,
                       unsigned* __restrict__ W1Kp, unsigned* __restrict__ W1Vp) {
  int idx = blockIdx.x * blockDim.x + threadIdx.x;
  if (idx < 2097152) {                       // 4x 1024x1024 weights -> wtile
    int seg = idx >> 19, local = idx & 524287;
    const float* s = seg == 0 ? Wq : seg == 1 ? Wk : seg == 2 ? Wv : Wo;
    unsigned* d = seg == 0 ? Wqp : seg == 1 ? Wkp : seg == 2 ? Wvp : Wop;
    int p = local & 7, half = (local >> 3) & 1, n = (local >> 4) & 1023, kb = local >> 14;
    int k = kb * 32 + half * 16 + 2 * p;
    d[local] = pack2(s[k * 1024 + n], s[(k + 1) * 1024 + n]);
  } else if (idx < 2097152 + 65536) {        // 2x 64x1024 weights -> wtile (kb=0..1)
    int t = idx - 2097152, seg = t >> 15, local = t & 32767;
    const float* s = seg ? W2V : W2K;
    unsigned* d = seg ? W2Vp : W2Kp;
    int p = local & 7, half = (local >> 3) & 1, n = (local >> 4) & 1023, kb = local >> 14;
    int k = kb * 32 + half * 16 + 2 * p;
    d[local] = pack2(s[k * 1024 + n], s[(k + 1) * 1024 + n]);
  } else if (idx < 2097152 + 65536 + 786432) {  // q/k/v activations
    int t = idx - (2097152 + 65536), seg = t >> 18, local = t & 262143;
    const float* s = seg == 0 ? Xq : seg == 1 ? Xk : Xv;
    unsigned* d = seg == 0 ? Xqd : seg == 1 ? Xkd : Xvd;
    d[local] = pack2(s[2 * local], s[2 * local + 1]);
  } else if (idx < 2097152 + 65536 + 786432 + 524288) {  // edges transpose
    int t = idx - (2097152 + 65536 + 786432);
    int row = t >> 3, p = t & 7;             // row = (b,i,j); 8 dwords of e
    int b = row >> 14, ij = row & 16383, i = ij >> 7, j = ij & 127;
    Ed[t] = pack2(edges[((b * 16 + 2 * p) * 128 + i) * 128 + j],
                  edges[((b * 16 + 2 * p + 1) * 128 + i) * 128 + j]);
  } else {                                   // W1 (16x64) x2 -> per-column tiles
    int t = idx - (2097152 + 65536 + 786432 + 524288);
    if (t >= 1024) return;
    int seg = t >> 9, local = t & 511;
    const float* s = seg ? W1V : W1K;
    unsigned* d = seg ? W1Vp : W1Kp;
    int p = local & 7, c = local >> 3;
    d[local] = pack2(s[(2 * p) * 64 + c], s[(2 * p + 1) * 64 + c]);
  }
}

// B operand from wtile weights: dwords base+0..3 and base+8..11 (A-style stride)
#define WTILE_B(Wp, kk, col) load16_astride(&(Wp)[(((kk) >> 5) * 1024 + (col)) * 16 + (akb >> 1)])

// ---------------------------------------------------------------------------
// 1) Projection: Y = X @ W + b (512x1024 @ 1024x1024), bf16 out, head-split.
//    Launched 3x with uniform pointer args. transposed=1 (v): store [b,h,dk,n].
__global__ void k_proj(const unsigned* __restrict__ Xd, const unsigned* __restrict__ Wp,
                       const float* __restrict__ bias, unsigned short* __restrict__ Y,
                       int transposed) {
  LANE_SETUP();
  int wid = WID();                     // 2048 tiles
  int r0 = (wid >> 6) << 4, c0 = (wid & 63) << 4;
  v8f acc = {};
  for (int kk = 0; kk < 1024; kk += 32) {
    if (kk + 32 < 1024)
      __builtin_prefetch(&Wp[(((kk + 32) >> 5) * 1024 + c0 + bn) * 16 + (bkb >> 1)], 0, 1);
    v16bf va = load16_astride(&Xd[((r0 + am) * 1024 + kk + akb) >> 1]);
    v16bf vb = WTILE_B(Wp, kk, c0 + bn);
    acc = wmma_bf16(va, vb, acc);
  }
  int c = c0 + bn, h = c >> 6, dk = c & 63;
  float bc = bias[c];
#pragma unroll
  for (int v = 0; v < 8; ++v) {
    int r = r0 + v + hi * 8, b = r >> 7, n = r & 127;
    unsigned short val = f2bf_bits(acc[v] + bc);
    if (transposed) Y[((b * 16 + h) * 64 + dk) * 128 + n] = val;
    else            Y[((b * 16 + h) * 128 + n) * 64 + dk] = val;
  }
}

// ---------------------------------------------------------------------------
// 2) Edge hidden: H[row=(b,i,j), c] = leakyrelu(Ebf[row,:] @ W1 + b1), K=16 padded.
//    Launched 2x. transposed=1 (V branch): store [(b,i), c, j]; else [row, c].
__global__ void k_hidden(const unsigned* __restrict__ Ed, const unsigned* __restrict__ W1p,
                         const float* __restrict__ b1, unsigned short* __restrict__ H,
                         int transposed) {
  LANE_SETUP();
  int wid = WID();                     // 16384 tiles
  int row0 = (wid >> 2) << 4, c0 = (wid & 3) << 4;
  int row = row0 + am;
  v8u ua = {}, ub = {};
  {
    int base = row * 8 + (akb >> 1);   // dwords of e: lanes 0-15 e=0..7, 16-31 e=8..15
#pragma unroll
    for (int p = 0; p < 4; ++p) ua[p] = Ed[base + p];
  }                                    // K in [16,32) stays zero
  {                                    // lane half h: e = 8h..8h+7; K in [16,32) stays zero
    const unsigned* w = &W1p[(c0 + bn) * 8 + (akb >> 1)];
#pragma unroll
    for (int p = 0; p < 4; ++p) ub[p] = w[p];
  }
  v8f acc = {};
  acc = wmma_bf16(__builtin_bit_cast(v16bf, ua), __builtin_bit_cast(v16bf, ub), acc);
  float bc = b1[c0 + bn];
#pragma unroll
  for (int v = 0; v < 8; ++v) {
    float x = acc[v] + bc;
    x = x > 0.f ? x : 0.1f * x;        // LeakyReLU(0.1)
    int rowv = row0 + v + hi * 8;
    unsigned short val = f2bf_bits(x);
    if (transposed) {
      int bb = rowv >> 14, ij2 = rowv & 16383, ii = ij2 >> 7, jj = ij2 & 127;
      H[((bb * 128 + ii) * 64 + c0 + bn) * 128 + jj] = val;
    } else {
      H[rowv * 64 + c0 + bn] = val;
    }
  }
}

// ---------------------------------------------------------------------------
// 3a) u2[h,c] = sum_d eK_W2[c, h*64+d] * rel_v[h,d]
__global__ void k_u2(const float* __restrict__ W2K, const float* __restrict__ relv,
                     float* __restrict__ u2) {
  int idx = blockIdx.x * blockDim.x + threadIdx.x;
  if (idx >= 1024) return;
  int h = idx >> 6, c = idx & 63;
  float s = 0.f;
  for (int d = 0; d < 64; ++d) s += W2K[c * 1024 + h * 64 + d] * relv[h * 64 + d];
  u2[idx] = s;
}

// 3b) qks = q + k (bf16, dword-wise), done once so k_gmod is pure loads
__global__ void k_qks(const unsigned* __restrict__ qd, const unsigned* __restrict__ kd,
                      unsigned* __restrict__ qks) {
  int idx = blockIdx.x * blockDim.x + threadIdx.x;  // 262144 dwords
  qks[idx] = addpack_bf(qd[idx], kd[idx]);
}

// 3c) Gmod[b,h,i,c] = inv * sum_d (q+k)[b,h,i,d] * eK_W2[c,h*64+d] + u2[h,c]
__global__ void k_gmod(const unsigned* __restrict__ qks, const unsigned* __restrict__ W2Kp,
                       const float* __restrict__ u2, unsigned short* __restrict__ Gm) {
  LANE_SETUP();
  int wid = WID();
  int bh = wid >> 5, rem = wid & 31;
  int i0 = (rem >> 2) << 4, c0 = (rem & 3) << 4;
  int h = bh & 15;
  v8f acc = {};
#pragma unroll
  for (int kk = 0; kk < 64; kk += 32) {
    v16bf va = load16_astride(&qks[((bh * 128 + i0 + am) * 64 + kk + akb) >> 1]);
    v16bf vb = WTILE_B(W2Kp, kk, h * 64 + c0 + bn);
    acc = wmma_bf16(va, vb, acc);
  }
  float uu = u2[h * 64 + c0 + bn];
#pragma unroll
  for (int v = 0; v < 8; ++v)
    Gm[(bh * 128 + i0 + v + hi * 8) * 64 + c0 + bn] = f2bf_bits(0.125f * acc[v] + uu);
}

// ---------------------------------------------------------------------------
// 4) S1[b,h,i,j] = inv * q . k  (per-head 128x128x64 GEMM, pure dword loads)
__global__ void k_s1(const unsigned short* __restrict__ qh, const unsigned short* __restrict__ kh,
                     float* __restrict__ S1) {
  LANE_SETUP();
  int wid = WID();
  int bh = wid >> 6, rem = wid & 63;
  int i0 = (rem >> 3) << 4, j0 = (rem & 7) << 4;
  const unsigned* qd = (const unsigned*)qh;
  const unsigned* kd = (const unsigned*)kh;
  v8f acc = {};
#pragma unroll
  for (int kk = 0; kk < 64; kk += 32) {
    v16bf va = load16_astride(&qd[((bh * 128 + i0 + am) * 64 + kk + akb) >> 1]);
    v16bf vb = load16_astride(&kd[((bh * 128 + j0 + bn) * 64 + kk + akb) >> 1]);
    acc = wmma_bf16(va, vb, acc);
  }
#pragma unroll
  for (int v = 0; v < 8; ++v)
    S1[(bh * 128 + i0 + v + hi * 8) * 128 + j0 + bn] = 0.125f * acc[v];
}

// ---------------------------------------------------------------------------
// 5) Per-(b,i): S = S1 + Gmod @ hiddenK^T, softmax over j, T = P @ hiddenV
__global__ void k_att(const unsigned short* __restrict__ Gm, const unsigned short* __restrict__ Hk,
                      const unsigned short* __restrict__ Hvt, const float* __restrict__ S1,
                      unsigned short* __restrict__ P, unsigned short* __restrict__ Tb) {
  __shared__ float Sm[16 * 128];
  __shared__ unsigned Pl[16 * 64];     // bf16 pairs: row h, 64 dwords of j
  LANE_SETUP();
  const int w = threadIdx.x >> 5;
  const int b = blockIdx.x >> 7, i = blockIdx.x & 127;
  const unsigned* gd = (const unsigned*)Gm;
  const unsigned* hkd = (const unsigned*)Hk;
  const unsigned* hvd = (const unsigned*)Hvt;
  unsigned* Pd = (unsigned*)P;

  // --- scores: S[h, j] = S1 + sum_c Gmod[b,h,i,c] * hiddenK[b,i,j,c]
#pragma unroll
  for (int t = 0; t < 2; ++t) {
    int j0 = (w * 2 + t) * 16;
    v8f acc = {};
#pragma unroll
    for (int kk = 0; kk < 64; kk += 32) {
      v16bf va = load16_astride(&gd[(((b * 16 + am) * 128 + i) * 64 + kk + akb) >> 1]);
      v16bf vb = load16_astride(&hkd[(((b * 128 + i) * 128 + j0 + bn) * 64 + kk + akb) >> 1]);
      acc = wmma_bf16(va, vb, acc);
    }
#pragma unroll
    for (int v = 0; v < 8; ++v) {
      int h = v + hi * 8;
      Sm[h * 128 + j0 + bn] = acc[v] + S1[((b * 16 + h) * 128 + i) * 128 + j0 + bn];
    }
  }
  __syncthreads();

  // --- softmax over j (row-constant terms cancel -> omitted)
#pragma unroll
  for (int rr = 0; rr < 4; ++rr) {
    int r = w + rr * 4, col = lane * 4;
    float x0 = Sm[r * 128 + col], x1 = Sm[r * 128 + col + 1];
    float x2 = Sm[r * 128 + col + 2], x3 = Sm[r * 128 + col + 3];
    float mx = fmaxf(fmaxf(x0, x1), fmaxf(x2, x3));
#pragma unroll
    for (int off = 16; off > 0; off >>= 1) mx = fmaxf(mx, __shfl_xor(mx, off, 32));
    float e0 = __expf(x0 - mx), e1 = __expf(x1 - mx), e2 = __expf(x2 - mx), e3 = __expf(x3 - mx);
    float sm = e0 + e1 + e2 + e3;
#pragma unroll
    for (int off = 16; off > 0; off >>= 1) sm += __shfl_xor(sm, off, 32);
    float rinv = 1.0f / sm;
    unsigned dw0 = pack2(e0 * rinv, e1 * rinv);
    unsigned dw1 = pack2(e2 * rinv, e3 * rinv);
    Pl[r * 64 + lane * 2] = dw0;
    Pl[r * 64 + lane * 2 + 1] = dw1;
    int pg = ((b * 16 + r) * 128 + i) * 64 + lane * 2;
    Pd[pg] = dw0; Pd[pg + 1] = dw1;
  }
  __syncthreads();

  // --- T[h, c] = sum_j P[h,j] * hiddenV[b,i,j,c]  (B from transposed hiddenV)
  int c0 = w * 16;
  v8f acc = {};
#pragma unroll
  for (int kk = 0; kk < 128; kk += 32) {
    v16bf va = load16_astride(&Pl[(am * 128 + kk + akb) >> 1]);
    v16bf vb = load16_astride(&hvd[(((b * 128 + i) * 64 + c0 + bn) * 128 + kk + akb) >> 1]);
    acc = wmma_bf16(va, vb, acc);
  }
#pragma unroll
  for (int v = 0; v < 8; ++v)
    Tb[((b * 16 + v + hi * 8) * 128 + i) * 64 + c0 + bn] = f2bf_bits(acc[v]);
}

// ---------------------------------------------------------------------------
// 6) AF[b,i,h*64+d] = (P @ V)[i,d] + (T @ eV_W2_h)[i,d] + eV_b2[h*64+d]
__global__ void k_af(const unsigned short* __restrict__ P, const unsigned short* __restrict__ vt,
                     const unsigned short* __restrict__ Tb, const unsigned* __restrict__ W2Vp,
                     const float* __restrict__ b2V, unsigned short* __restrict__ AF) {
  LANE_SETUP();
  int wid = WID();
  int bh = wid >> 5, rem = wid & 31;
  int i0 = (rem >> 2) << 4, d0 = (rem & 3) << 4;
  int b = bh >> 4, h = bh & 15;
  const unsigned* pd = (const unsigned*)P;
  const unsigned* vd = (const unsigned*)vt;
  const unsigned* td = (const unsigned*)Tb;
  v8f acc = {};
#pragma unroll
  for (int kk = 0; kk < 128; kk += 32) {    // P @ V  (V transposed: [bh, d, n])
    v16bf va = load16_astride(&pd[((bh * 128 + i0 + am) * 128 + kk + akb) >> 1]);
    v16bf vb = load16_astride(&vd[((bh * 64 + d0 + bn) * 128 + kk + akb) >> 1]);
    acc = wmma_bf16(va, vb, acc);
  }
#pragma unroll
  for (int kk = 0; kk < 64; kk += 32) {     // T @ eV_W2_h
    v16bf va = load16_astride(&td[((bh * 128 + i0 + am) * 64 + kk + akb) >> 1]);
    v16bf vb = WTILE_B(W2Vp, kk, h * 64 + d0 + bn);
    acc = wmma_bf16(va, vb, acc);
  }
  float bias = b2V[h * 64 + d0 + bn];       // softmax rows sum to 1 -> + b2v
#pragma unroll
  for (int v = 0; v < 8; ++v)
    AF[(b * 128 + i0 + v + hi * 8) * 1024 + h * 64 + d0 + bn] = f2bf_bits(acc[v] + bias);
}

// ---------------------------------------------------------------------------
// 7) out = AF @ Wo + bo  (512x1024 @ 1024x1024, fp32 out)
__global__ void k_out(const unsigned short* __restrict__ AF, const unsigned* __restrict__ Wop,
                      const float* __restrict__ bo, float* __restrict__ out) {
  LANE_SETUP();
  int wid = WID();
  int r0 = (wid >> 6) << 4, c0 = (wid & 63) << 4;
  const unsigned* ad = (const unsigned*)AF;
  v8f acc = {};
  for (int kk = 0; kk < 1024; kk += 32) {
    if (kk + 32 < 1024)
      __builtin_prefetch(&Wop[(((kk + 32) >> 5) * 1024 + c0 + bn) * 16 + (bkb >> 1)], 0, 1);
    v16bf va = load16_astride(&ad[((r0 + am) * 1024 + kk + akb) >> 1]);
    v16bf vb = WTILE_B(Wop, kk, c0 + bn);
    acc = wmma_bf16(va, vb, acc);
  }
  float bc = bo[c0 + bn];
#pragma unroll
  for (int v = 0; v < 8; ++v)
    out[(r0 + v + hi * 8) * 1024 + c0 + bn] = acc[v] + bc;
}

// ---------------------------------------------------------------------------
extern "C" void kernel_launch(void* const* d_in, const int* in_sizes, int n_in,
                              void* d_out, int out_size, void* d_ws, size_t ws_size,
                              hipStream_t stream) {
  const float* query = (const float*)d_in[0];
  const float* key   = (const float*)d_in[1];
  const float* value = (const float*)d_in[2];
  const float* edges = (const float*)d_in[3];
  const float* Wq = (const float*)d_in[4];  const float* bq = (const float*)d_in[5];
  const float* Wk = (const float*)d_in[6];  const float* bk = (const float*)d_in[7];
  const float* Wv = (const float*)d_in[8];  const float* bv = (const float*)d_in[9];
  const float* Wo = (const float*)d_in[10]; const float* bo = (const float*)d_in[11];
  const float* eK_W1 = (const float*)d_in[12]; const float* eK_b1 = (const float*)d_in[13];
  const float* eK_W2 = (const float*)d_in[14]; /* eK_b2 (d_in[15]) cancels in softmax */
  const float* eV_W1 = (const float*)d_in[16]; const float* eV_b1 = (const float*)d_in[17];
  const float* eV_W2 = (const float*)d_in[18]; const float* eV_b2 = (const float*)d_in[19];
  /* rel_u (d_in[20]) cancels in softmax */
  const float* rel_v = (const float*)d_in[21];
  float* out = (float*)d_out;
  (void)in_sizes; (void)n_in; (void)out_size; (void)ws_size;

  // workspace layout
  unsigned short* qh  = (unsigned short*)d_ws;         // [B,H,N,DK]    524288
  unsigned short* kh  = qh + 524288;
  unsigned short* vt  = kh + 524288;                   // [B,H,DK,N]    524288
  unsigned short* Hk  = vt + 524288;                   // [B,N,N,DK]   4194304
  unsigned short* Hvt = Hk + 4194304;                  // [B,N,DK,N]   4194304
  unsigned short* Gm  = Hvt + 4194304;                 // [B,H,N,DK]    524288
  unsigned short* P   = Gm + 524288;                   // [B,H,N,N]    1048576
  unsigned short* Tb  = P  + 1048576;                  // [B,H,N,DK]    524288
  unsigned short* AF  = Tb + 524288;                   // [B,N,D]       524288
  float* u2 = (float*)(AF + 524288);                   // [H,DK]           1024
  float* S1 = u2 + 1024;                               // [B,H,N,N]    1048576
  unsigned* Wqp  = (unsigned*)(S1 + 1048576);          // wtile weights
  unsigned* Wkp  = Wqp + 524288;
  unsigned* Wvp  = Wkp + 524288;
  unsigned* Wop  = Wvp + 524288;
  unsigned* W2Kp = Wop + 524288;                       // 32768
  unsigned* W2Vp = W2Kp + 32768;                       // 32768
  unsigned* Xqd  = W2Vp + 32768;                       // bf16 activations 262144 each
  unsigned* Xkd  = Xqd + 262144;
  unsigned* Xvd  = Xkd + 262144;
  unsigned* Ed   = Xvd + 262144;                       // edges [b,i,j,e]  524288
  unsigned* W1Kp = Ed + 524288;                        // 512
  unsigned* W1Vp = W1Kp + 512;                         // 512
  unsigned* qks  = W1Vp + 512;                         // q+k bf16         262144

  k_prep  <<<13572, 256, 0, stream>>>(Wq, Wk, Wv, Wo, eK_W2, eV_W2, query, key, value,
                                      edges, eK_W1, eV_W1,
                                      Wqp, Wkp, Wvp, Wop, W2Kp, W2Vp,
                                      Xqd, Xkd, Xvd, Ed, W1Kp, W1Vp);
  k_u2    <<<4,    256, 0, stream>>>(eK_W2, rel_v, u2);
  k_proj  <<<256,  256, 0, stream>>>(Xqd, Wqp, bq, qh, 0);
  k_proj  <<<256,  256, 0, stream>>>(Xkd, Wkp, bk, kh, 0);
  k_proj  <<<256,  256, 0, stream>>>(Xvd, Wvp, bv, vt, 1);
  k_qks   <<<1024, 256, 0, stream>>>((const unsigned*)qh, (const unsigned*)kh, qks);
  k_hidden<<<2048, 256, 0, stream>>>(Ed, W1Kp, eK_b1, Hk, 0);
  k_hidden<<<2048, 256, 0, stream>>>(Ed, W1Vp, eV_b1, Hvt, 1);
  k_gmod  <<<256,  256, 0, stream>>>(qks, W2Kp, u2, Gm);
  k_s1    <<<512,  256, 0, stream>>>(qh, kh, S1);
  k_att   <<<512,  128, 0, stream>>>(Gm, Hk, Hvt, S1, P, Tb);
  k_af    <<<256,  256, 0, stream>>>(P, vt, Tb, W2Vp, eV_b2, AF);
  k_out   <<<256,  256, 0, stream>>>(AF, Wop, bo, out);
}
